// BidirectionalLSTM_3023656976980
// MI455X (gfx1250) — matrix operand order repaired and checked by reading, not compiled, not run on hardware
//
#include <hip/hip_runtime.h>
#include <hip/hip_bf16.h>

typedef __attribute__((ext_vector_type(16))) __bf16 v16bf;
typedef __attribute__((ext_vector_type(8)))  float  v8f;
typedef unsigned int v4u __attribute__((ext_vector_type(4)));
typedef int v8i __attribute__((ext_vector_type(8)));
typedef int v4i __attribute__((ext_vector_type(4)));

#if defined(__has_builtin)
#if __has_builtin(__builtin_amdgcn_tensor_load_to_lds)
#define USE_TDM_B 1
#endif
#endif

union FragBF {
    v16bf v;
    unsigned short s[16];
    uint2 q[4];
    uint4 o[2];
};

static __device__ __forceinline__ unsigned short f2bf(float f) {
    unsigned u = __float_as_uint(f);
    u += 0x7FFFu + ((u >> 16) & 1u);          // round-to-nearest-even
    return (unsigned short)(u >> 16);
}

static __device__ __forceinline__ v8f wmma_bf16(v16bf a, v16bf b, v8f c) {
    return __builtin_amdgcn_wmma_f32_16x16x32_bf16(
        false, a, false, b, (short)0, c, false, false);
}

static __device__ __forceinline__ float sigf(float x) {
    return 1.0f / (1.0f + __expf(-x));
}

static __device__ __forceinline__ void async_copy_b128(unsigned ldsOff,
                                                       const void* gptr) {
    asm volatile("global_load_async_to_lds_b128 %0, %1, off"
                 :: "v"(ldsOff), "v"((unsigned long long)(uintptr_t)gptr)
                 : "memory");
}
static __device__ __forceinline__ void wait_async0() {
    asm volatile("s_wait_asynccnt 0x0" ::: "memory");
}

// ---------------------------------------------------------------------------
// One-time conversions (weights reused 2048x -> convert once, move bf16 after)
// ---------------------------------------------------------------------------
__global__ __launch_bounds__(256) void conv_lin(const float* __restrict__ src,
                                                unsigned short* __restrict__ dst) {
    int i = (blockIdx.x * 256 + threadIdx.x) * 4;
    float4 f = *(const float4*)(src + i);
    uint2 p;
    p.x = (unsigned)f2bf(f.x) | ((unsigned)f2bf(f.y) << 16);
    p.y = (unsigned)f2bf(f.z) | ((unsigned)f2bf(f.w) << 16);
    *(uint2*)(dst + i) = p;
}

// src: K x N fp32 row-major  ->  dst: N x K bf16 (transposed)
__global__ __launch_bounds__(256) void conv_t(const float* __restrict__ src,
                                              unsigned short* __restrict__ dst,
                                              int K, int N) {
    int i = blockIdx.x * 256 + threadIdx.x;   // over K*N
    int k = i / N, n = i % N;
    dst[(long)n * K + k] = f2bf(src[i]);
}

__global__ void init_ctr(unsigned int* c) {
    if (threadIdx.x < 4) c[threadIdx.x] = 0u;
}

// ---------------------------------------------------------------------------
// pre = A(M x K bf16, gathered rows) @ Wt(N x K bf16, pre-transposed) + bias
// Block tile 64(M) x 128(N); 8 waves; async/TDM staging into padded LDS.
// ---------------------------------------------------------------------------
#define GBM 64
#define GBN 128
#define GBK 32
#define TPITCH 40     // 32 data + 8 pad ushorts = 80B (16B aligned, conflict-free)

__global__ __launch_bounds__(256) void gemm_pre(
    const unsigned short* __restrict__ A, const unsigned short* __restrict__ Wt,
    const float* __restrict__ bias, float* __restrict__ C,
    int N, int K, long long strideT, long long strideB, int Bsz)
{
    __shared__ unsigned short Alds[GBM * TPITCH];
    __shared__ unsigned short Blds[GBN * TPITCH];

    const int tid   = threadIdx.x;
    const int wave  = tid >> 5;
    const int lane  = tid & 31;
    const int lm    = lane & 15;
    const int half  = lane >> 4;
    const int mtile = wave & 3;
    const int nhalf = wave >> 2;
    const int blockM = blockIdx.x * GBM;
    const int blockN = blockIdx.y * GBN;

    // per-thread A staging slot: row r (0..63), 16B quarter q (0..3)
    const int rA = tid >> 2, qA = tid & 3;
    {
    }
    const int grow = blockM + rA;
    const int tt = grow / Bsz, bb = grow % Bsz;
    const unsigned short* arow = A + (long long)tt * strideT + (long long)bb * strideB + qA * 8;
    const unsigned aOff = (unsigned)(uintptr_t)&Alds[rA * TPITCH + qA * 8];

#ifndef USE_TDM_B
    // per-thread B staging: 2 chunks of 16B each
    const int cid0 = tid * 2;
    const int nB0 = cid0 >> 2, qB0 = cid0 & 3;
    const int nB1 = (cid0 + 1) >> 2, qB1 = (cid0 + 1) & 3;
    const unsigned short* brow0 = Wt + (long long)(blockN + nB0) * K + qB0 * 8;
    const unsigned short* brow1 = Wt + (long long)(blockN + nB1) * K + qB1 * 8;
    const unsigned bOff0 = (unsigned)(uintptr_t)&Blds[nB0 * TPITCH + qB0 * 8];
    const unsigned bOff1 = (unsigned)(uintptr_t)&Blds[nB1 * TPITCH + qB1 * 8];
#endif

    v8f acc[4] = {};

    for (int k0 = 0; k0 < K; k0 += GBK) {
        // ---- A tile: 64 rows x 32 bf16, per-lane async 16B copies ----
        async_copy_b128(aOff, arow + k0);

#ifdef USE_TDM_B
        // ---- B tile: one TDM descriptor (128 x 32 bf16 2D tile, row stride K,
        //      LDS padding 4 dwords after every 16 dwords -> TPITCH rows) ----
        if (wave == 0) {
            unsigned long long ga =
                (unsigned long long)(uintptr_t)(Wt + (long long)blockN * K + k0);
            unsigned ldsB = (unsigned)(uintptr_t)&Blds[0];
            v4u g0 = { 1u,                         // count=1, user descriptor
                       ldsB,                        // lds_addr
                       (unsigned)ga,                // global_addr lo
                       (unsigned)((ga >> 32) & 0x1ffffffULL) | 0x80000000u }; // type=2
            v8i g1 = { (int)((1u << 16)  |         // data_size = 2B
                             (1u << 20)  |         // pad_enable
                             (3u << 22)  |         // pad_interval: 16 dwords
                             (3u << 25)),          // pad_amount: 4 dwords
                       (int)(32u << 16),           // tensor_dim0 = 32
                       (int)(128u << 16),          // tensor_dim1 = 128
                       (int)(32u << 16),           // tile_dim0 = 32
                       128,                        // tile_dim1 = 128
                       K, 0, 0 };                  // tensor_dim0_stride = K
            v4i gz = { 0, 0, 0, 0 };
#if __has_include(<hip/amd_detail/amd_gfx1250_TDM.h>)
            v8i gz8 = { 0, 0, 0, 0, 0, 0, 0, 0 };
            __builtin_amdgcn_tensor_load_to_lds(g0, g1, gz, gz, gz8, 0);
#else
            __builtin_amdgcn_tensor_load_to_lds(g0, g1, gz, gz, 0);
#endif
        }
#else
        async_copy_b128(bOff0, brow0 + k0);
        async_copy_b128(bOff1, brow1 + k0);
#endif
        wait_async0();
#ifdef USE_TDM_B
        if (wave == 0) __builtin_amdgcn_s_wait_tensorcnt(0);
#endif
        __syncthreads();

        // ---- fragments: aligned ds_load_b128, conflict-free (80B pitch) ----
        FragBF af;
        {
            const unsigned short* rowA = &Alds[(mtile * 16 + lm) * TPITCH];
            af.o[0] = *(const uint4*)(rowA + half * 8);
            af.o[1] = *(const uint4*)(rowA + 16 + half * 8);
        }
#pragma unroll
        for (int nt = 0; nt < 4; ++nt) {
            FragBF bf;
            const unsigned short* rowB = &Blds[(nhalf * 64 + nt * 16 + lm) * TPITCH];
            bf.o[0] = *(const uint4*)(rowB + half * 8);
            bf.o[1] = *(const uint4*)(rowB + 16 + half * 8);
            acc[nt] = wmma_bf16(af.v, bf.v, acc[nt]);
        }
        __syncthreads();
    }

#pragma unroll
    for (int nt = 0; nt < 4; ++nt) {
        int n = blockN + nhalf * 64 + nt * 16 + lm;
        float bv = bias[n];
#pragma unroll
        for (int r = 0; r < 8; ++r) {
            int m = blockM + mtile * 16 + r + 8 * half;
            C[(long long)m * N + n] = acc[nt][r] + bv;
        }
    }
}

// ---------------------------------------------------------------------------
// LSTM recurrence: 16 cooperating WGs per direction; h kept bf16 end-to-end.
// ---------------------------------------------------------------------------
#define RB 8
#define RH 512
#define RG 2048
#define NSLICE 16
#define SC 32
#define HP2 544          // 512 + 32 pad ushorts; 1088B row pitch (16B aligned)

__global__ __launch_bounds__(256) void lstm_rec(
    const float* __restrict__ pre_f, const float* __restrict__ pre_b,
    const unsigned short* __restrict__ wht_f, const unsigned short* __restrict__ wht_b,
    unsigned short* __restrict__ hbuf,     // [2][8][512] bf16
    unsigned int* __restrict__ counters,   // [2]
    unsigned short* __restrict__ outb,     // layer0: y0 (T,B,2H) bf16
    float* __restrict__ outf,              // layer1: d_out (B,T,2H) fp32
    int btMajor, int T)
{
    const int dir = blockIdx.y;
    const float* pre          = dir ? pre_b : pre_f;
    const unsigned short* wht = dir ? wht_b : wht_f;
    unsigned short* hb        = hbuf + dir * (RB * RH);
    unsigned int* ctr         = counters + dir;
    const int slice           = blockIdx.x;

    __shared__ unsigned short h_lds[RB * HP2];   // full h, bf16
    __shared__ float c_lds[RB * SC];
    __shared__ float z_lds[4 * RB * SC];

    const int tid  = threadIdx.x;
    const int wave = tid >> 5;
    const int lane = tid & 31;
    const int lm   = lane & 15;
    const int half = lane >> 4;
    const int gate = wave >> 1;
    const int colbase = gate * RH + slice * SC + (wave & 1) * 16;

    for (int i = tid; i < RB * HP2; i += 256) h_lds[i] = 0;
    if (tid < RB * SC) c_lds[tid] = 0.0f;
    __syncthreads();

    const unsigned short* wcol = wht + (long)(colbase + lm) * RH;
    const bool mv = (lm < RB);
    const unsigned short* hrow = &h_lds[(lm & 7) * HP2];
    const uint4 z4 = { 0u, 0u, 0u, 0u };

    for (int s = 0; s < T; ++s) {
        const int t = dir ? (T - 1 - s) : s;

        // ---- z-tile = h @ Wh[:, col..col+16), K = 512 : 16 WMMAs ----
        v8f acc = {};
#pragma unroll 4
        for (int kk = 0; kk < 16; ++kk) {
            const int kb = kk * 32;
            FragBF a, b;
            const unsigned short* hr = hrow + kb + half * 8;
            a.o[0] = mv ? *(const uint4*)(hr)      : z4;
            a.o[1] = mv ? *(const uint4*)(hr + 16) : z4;
            const unsigned short* wr = wcol + kb + half * 8;
            b.o[0] = *(const uint4*)(wr);
            b.o[1] = *(const uint4*)(wr + 16);
            acc = wmma_bf16(a.v, b.v, acc);
        }

        // ---- add input projection, park gates in LDS ----
        {
            const long prow = (long)t * RB;
#pragma unroll
            for (int r = 0; r < 8; ++r) {
                int m = r + 8 * half;
                if (m < RB) {
                    float z = acc[r] + pre[(prow + m) * (long)RG + colbase + lm];
                    z_lds[gate * (RB * SC) + m * SC + (wave & 1) * 16 + lm] = z;
                }
            }
        }
        __syncthreads();

        // ---- elementwise cell update, 8x32 slice ----
        {
            int m   = tid >> 5;
            int col = tid & 31;
            int zi  = m * SC + col;
            float iv = z_lds[0 * (RB * SC) + zi];
            float fv = z_lds[1 * (RB * SC) + zi];
            float gv = z_lds[2 * (RB * SC) + zi];
            float ov = z_lds[3 * (RB * SC) + zi];
            float c = sigf(fv) * c_lds[zi] + sigf(iv) * tanhf(gv);
            float h = sigf(ov) * tanhf(c);
            c_lds[zi] = c;
            int n = slice * SC + col;
            hb[m * RH + n] = f2bf(h);
            if (btMajor)
                outf[((long)m * T + t) * 1024 + dir * RH + n] = h;
            else
                outb[((long)t * RB + m) * 1024 + dir * RH + n] = f2bf(h);
        }
        __threadfence();
        __syncthreads();

        // ---- device-scope barrier across the 16 slices of this direction ----
        if (tid == 0) {
            __hip_atomic_fetch_add(ctr, 1u, __ATOMIC_RELEASE, __HIP_MEMORY_SCOPE_AGENT);
            unsigned int target = (unsigned int)NSLICE * (unsigned int)(s + 1);
            while (__hip_atomic_load(ctr, __ATOMIC_ACQUIRE, __HIP_MEMORY_SCOPE_AGENT) < target)
                __builtin_amdgcn_s_sleep(2);
        }
        __syncthreads();

        // ---- refresh full h (8KB bf16) into LDS ----
        {
            int fi = tid * 16;               // 0..4095
            int m = fi >> 9, n = fi & 511;
            *(uint4*)&h_lds[m * HP2 + n]     = *(const uint4*)(hb + fi);
            *(uint4*)&h_lds[m * HP2 + n + 8] = *(const uint4*)(hb + fi + 8);
        }
        __syncthreads();
    }
}

// ---------------------------------------------------------------------------
extern "C" void kernel_launch(void* const* d_in, const int* in_sizes, int n_in,
                              void* d_out, int out_size, void* d_ws, size_t ws_size,
                              hipStream_t stream) {
    const int B = 8, T = 2048, D = 512, H = 512, G = 4 * H;
    const int M = T * B;

    const float* x    = (const float*)d_in[0];
    const float* Wx0f = (const float*)d_in[1];
    const float* Wh0f = (const float*)d_in[2];
    const float* b0f  = (const float*)d_in[3];
    const float* Wx0b = (const float*)d_in[4];
    const float* Wh0b = (const float*)d_in[5];
    const float* b0b  = (const float*)d_in[6];
    const float* Wx1f = (const float*)d_in[7];
    const float* Wh1f = (const float*)d_in[8];
    const float* b1f  = (const float*)d_in[9];
    const float* Wx1b = (const float*)d_in[10];
    const float* Wh1b = (const float*)d_in[11];
    const float* b1b  = (const float*)d_in[12];

    // ---- workspace layout (all bf16 operands for WMMA paths) ----
    char* ws = (char*)d_ws;
    size_t off = 0;
    unsigned short* xbf = (unsigned short*)(ws + off);  off += (size_t)B * T * D * 2;      // 16MB
    unsigned short* wxt0f = (unsigned short*)(ws + off); off += (size_t)G * D * 2;         // 2MB
    unsigned short* wxt0b = (unsigned short*)(ws + off); off += (size_t)G * D * 2;
    unsigned short* wxt1f = (unsigned short*)(ws + off); off += (size_t)G * 2 * H * 2;     // 4MB
    unsigned short* wxt1b = (unsigned short*)(ws + off); off += (size_t)G * 2 * H * 2;
    unsigned short* wht0f = (unsigned short*)(ws + off); off += (size_t)G * H * 2;         // 2MB
    unsigned short* wht0b = (unsigned short*)(ws + off); off += (size_t)G * H * 2;
    unsigned short* wht1f = (unsigned short*)(ws + off); off += (size_t)G * H * 2;
    unsigned short* wht1b = (unsigned short*)(ws + off); off += (size_t)G * H * 2;
    float* pre_f = (float*)(ws + off);                   off += (size_t)M * G * 4;         // 128MB
    float* pre_b = (float*)(ws + off);                   off += (size_t)M * G * 4;
    unsigned short* y0 = (unsigned short*)(ws + off);    off += (size_t)M * 2 * H * 2;     // 32MB
    unsigned short* hbuf = (unsigned short*)(ws + off);  off += (size_t)2 * B * H * 2;
    unsigned int* ctr = (unsigned int*)(ws + off);

    init_ctr<<<1, 32, 0, stream>>>(ctr);

    // ---- one-time bf16 conversions ----
    conv_lin<<<(B * T * D) / 1024, 256, 0, stream>>>(x, xbf);
    conv_t<<<(D * G) / 256, 256, 0, stream>>>(Wx0f, wxt0f, D, G);
    conv_t<<<(D * G) / 256, 256, 0, stream>>>(Wx0b, wxt0b, D, G);
    conv_t<<<(2 * H * G) / 256, 256, 0, stream>>>(Wx1f, wxt1f, 2 * H, G);
    conv_t<<<(2 * H * G) / 256, 256, 0, stream>>>(Wx1b, wxt1b, 2 * H, G);
    conv_t<<<(H * G) / 256, 256, 0, stream>>>(Wh0f, wht0f, H, G);
    conv_t<<<(H * G) / 256, 256, 0, stream>>>(Wh0b, wht0b, H, G);
    conv_t<<<(H * G) / 256, 256, 0, stream>>>(Wh1f, wht1f, H, G);
    conv_t<<<(H * G) / 256, 256, 0, stream>>>(Wh1b, wht1b, H, G);

    dim3 ggrid(M / GBM, G / GBN);   // 256 x 16

    // ---- layer 0: x is (B,T,D) bf16, row r=(t*B+b) -> t*D + b*(T*D) ----
    gemm_pre<<<ggrid, 256, 0, stream>>>(xbf, wxt0f, b0f, pre_f, G, D,
                                        (long long)D, (long long)T * D, B);
    gemm_pre<<<ggrid, 256, 0, stream>>>(xbf, wxt0b, b0b, pre_b, G, D,
                                        (long long)D, (long long)T * D, B);
    lstm_rec<<<dim3(NSLICE, 2), 256, 0, stream>>>(pre_f, pre_b, wht0f, wht0b,
                                                  hbuf, ctr + 0, y0, nullptr,
                                                  /*btMajor=*/0, T);

    // ---- layer 1: y0 is (T,B,2H) bf16, row r -> t*(B*2H) + b*(2H) ----
    gemm_pre<<<ggrid, 256, 0, stream>>>(y0, wxt1f, b1f, pre_f, G, 2 * H,
                                        (long long)B * 2 * H, (long long)2 * H, B);
    gemm_pre<<<ggrid, 256, 0, stream>>>(y0, wxt1b, b1b, pre_b, G, 2 * H,
                                        (long long)B * 2 * H, (long long)2 * H, B);
    lstm_rec<<<dim3(NSLICE, 2), 256, 0, stream>>>(pre_f, pre_b, wht1f, wht1b,
                                                  hbuf, ctr + 2, nullptr,
                                                  (float*)d_out, /*btMajor=*/1, T);
}
